// FEGM_57543971832251
// MI455X (gfx1250) — compile-verified, matrix operand order checked by reading, not run on hardware
//
#include <hip/hip_runtime.h>
#include <hip/hip_bf16.h>
#include <math.h>

#define DEVINL __device__ __forceinline__

typedef __bf16 bf16_t;
typedef __attribute__((ext_vector_type(16))) __bf16 v16bf;
typedef __attribute__((ext_vector_type(8)))  __bf16 v8bf;
typedef __attribute__((ext_vector_type(4)))  __bf16 v4bf;
typedef __attribute__((ext_vector_type(8)))  float  v8f;

#define BATCH 16
#define CCH   128
#define NHEAD 4
#define HDIM  32
#define KW    5
#define PSZ   12
#define GHW   8
#define NPATCH 64
#define FEAT  18432
#define NTOK  1024

// ---- WMMA operand loads (bf16, 16x16x32) ----------------------------------
// A 16x32 lane layout: lane m = lane&15, K-half kh = (lane&16)>>1 (0 or 8).
// The 16 bf16 elements are two contiguous runs: K in [kh,kh+8) and [16+kh,+8).
// => two 128-bit loads when source is row-major over K (stride % 8 == 0).
DEVINL v16bf load_row16(const bf16_t* p, int stride, int lane) {
  int m = lane & 15; int kh = (lane & 16) >> 1;
  const bf16_t* q = p + m * stride + kh;
  v8bf lo = *(const v8bf*)q;          // K kh..kh+7
  v8bf hi = *(const v8bf*)(q + 16);   // K 16+kh..16+kh+7
  return __builtin_shufflevector(lo, hi, 0,1,2,3,4,5,6,7,8,9,10,11,12,13,14,15);
}
#define load_A  load_row16   // p[m*stride + k]
#define load_BT load_row16   // p[n*stride + k] (K contiguous per N row)

DEVINL v8f wmma_bf16(v16bf a, v16bf b, v8f c) {
  return __builtin_amdgcn_wmma_f32_16x16x32_bf16(false, a, false, b, (short)0, c, false, false);
}
// 16x16 GEMM tile over K=128: A row-major [row][k] stride 128, W row-major [j][k].
DEVINL v8f gemm_tile_128(const bf16_t* Arow, const bf16_t* Wrow, int lane) {
  v8f acc = {};
#pragma unroll
  for (int s = 0; s < 4; ++s) {
    v16bf a = load_A(Arow + s * 32, 128, lane);
    v16bf b = load_BT(Wrow + s * 32, 128, lane);
    acc = wmma_bf16(a, b, acc);
  }
  return acc;
}

// ---- K0: weight conversion to bf16 ---------------------------------------
__global__ void k_convert(const float* __restrict__ qkv_w, const float* __restrict__ yv_w,
                          const float* __restrict__ proj_w, const float* __restrict__ c1d_w,
                          bf16_t* wq, bf16_t* wyv, bf16_t* wproj, bf16_t* wc) {
  int t = blockIdx.x * 256 + threadIdx.x;
  if (t < 384 * 128) wq[t] = (bf16_t)qkv_w[t];
  if (t < 128 * 128) { wyv[t] = (bf16_t)yv_w[t]; wproj[t] = (bf16_t)proj_w[t]; }
  if (t < 128 * 640) {                 // wc[c][tap*128+cin]
    int c = t / 640, k = t % 640, tap = k >> 7, cin = k & 127;
    wc[t] = (bf16_t)c1d_w[(c * 128 + cin) * 5 + tap];
  }
}

// ---- K1: patch-embed conv (5-tap shifted GEMM) + BN + ReLU + mask + pool --
__global__ __launch_bounds__(256) void k_patch(const float* __restrict__ s1,
    const float* __restrict__ pe_w, const float* __restrict__ bn_g,
    const float* __restrict__ bn_b, const float* __restrict__ bn_m,
    const float* __restrict__ bn_v, const int* __restrict__ index,
    float* __restrict__ shortbuf) {
  __shared__ __align__(16) bf16_t Wl[64 * 320];   // A: [p_out][kappa = tap*64 + p_in]
  __shared__ __align__(16) bf16_t Xl[132 * 72];   // B^T: [feat col 0..131][p_in], stride 72
  __shared__ float bnsc[64], bnsh[64];
  __shared__ int selm[64];
  int tid = threadIdx.x, lane = tid & 31, wid = tid >> 5;
  int b = blockIdx.x >> 4, pyx = blockIdx.x & 15;
  int py = pyx >> 2, px = pyx & 3;

  for (int t = tid; t < 64 * 320; t += 256) {
    int p = t / 320, k = t % 320;
    Wl[t] = (bf16_t)pe_w[(p * 64 + (k & 63)) * 5 + (k >> 6)];
  }
  if (tid < 64) {
    float sc = bn_g[tid] * rsqrtf(bn_v[tid] + 1e-5f);
    bnsc[tid] = sc; bnsh[tid] = bn_b[tid] - bn_m[tid] * sc;
    selm[tid] = 0;
  }
  __syncthreads();
  if (tid < 4) {
    int idx = (index[2 * tid] / PSZ) * GHW + (index[2 * tid + 1] / PSZ);
    selm[idx] = 1;
  }

  int m_t = wid & 3;                    // p_out tile
  int ntbase = (wid >> 2) * 4;          // 4 feat tiles of 16 (c dimension)
  v8f zero = {};
  v8f accP[4];
#pragma unroll
  for (int i = 0; i < 4; ++i) accP[i] = zero;

  for (int dy = 0; dy < 3; ++dy)
  for (int dx = 0; dx < 3; ++dx) {
    int p1 = py * 3 + dy, p2 = px * 3 + dx;
    int f0 = (p1 * PSZ + p2) * CCH;
    __syncthreads();
    // stage X transposed: Xl[col][pin], packed 4 p_in per store
    for (int t = tid; t < 132 * 16; t += 256) {
      int col = t >> 4, pin0 = (t & 15) * 4;
      int f = f0 - 2 + col;
      v4bf pk = {};
      if (f >= 0 && f < FEAT) {
        int c = f & 127, pp = f >> 7;
        int p1x = pp / PSZ, p2x = pp % PSZ;
        const float* sp = &s1[((b * CCH + c) * 96 + p1x) * 96 + p2x];
#pragma unroll
        for (int i = 0; i < 4; ++i) {
          int pin = pin0 + i, gh = pin >> 3, gw = pin & 7;
          pk[i] = (bf16_t)sp[gh * PSZ * 96 + gw * PSZ];
        }
      }
      *(v4bf*)&Xl[col * 72 + pin0] = pk;
    }
    __syncthreads();

    v8f accT[4];
#pragma unroll
    for (int i = 0; i < 4; ++i) accT[i] = zero;
    for (int s = 0; s < 10; ++s) {      // K = 5 taps * 64 p_in, steps of 32
      int tap = s >> 1, pin0 = (s & 1) * 32;
      v16bf a = load_A(&Wl[(m_t * 16) * 320 + tap * 64 + pin0], 320, lane);
#pragma unroll
      for (int i = 0; i < 4; ++i) {
        int nt = ntbase + i;
        v16bf bb = load_BT(&Xl[(nt * 16 + tap) * 72 + pin0], 72, lane);
        accT[i] = wmma_bf16(a, bb, accT[i]);
      }
    }
    int hi = (lane & 16) >> 1;
#pragma unroll
    for (int i = 0; i < 4; ++i)
#pragma unroll
      for (int r = 0; r < 8; ++r) {
        int pout = m_t * 16 + r + hi;
        float y = accT[i][r] * bnsc[pout] + bnsh[pout];
        y = fmaxf(y, 0.f);
        if (!selm[pout]) y = y * y;     // squared unless selected patch row
        accP[i][r] += y;
      }
  }
  int hi = (lane & 16) >> 1, nlo = lane & 15;
#pragma unroll
  for (int i = 0; i < 4; ++i) {
    int c = (ntbase + i) * 16 + nlo;
#pragma unroll
    for (int r = 0; r < 8; ++r) {
      int pout = m_t * 16 + r + hi;
      int gh = pout >> 3, gw = pout & 7;
      int oy = gh * 4 + py, ox = gw * 4 + px;
      shortbuf[((b * CCH + c) * 32 + oy) * 32 + ox] = accP[i][r] * (1.f / 9.f);
    }
  }
}

// ---- K2a: 2x2 pool of o + LayerNorm -> of (bf16 [B,N,C]) ------------------
__global__ __launch_bounds__(256) void k_pool_ln_o(const float* __restrict__ o,
    const float* __restrict__ g, const float* __restrict__ bta, bf16_t* __restrict__ of) {
  int idx = blockIdx.x * 8 + (threadIdx.x >> 5);
  int lane = threadIdx.x & 31;
  int b = idx >> 10, n = idx & 1023;
  int oy = n >> 5, ox = n & 31;
  float vals[4], s = 0.f, s2 = 0.f;
#pragma unroll
  for (int i = 0; i < 4; ++i) {
    int c = lane + i * 32;
    const float* p = &o[((b * CCH + c) * 64 + oy * 2) * 64 + ox * 2];
    float v = 0.25f * (p[0] + p[1] + p[64] + p[65]);
    vals[i] = v; s += v; s2 += v * v;
  }
#pragma unroll
  for (int sh = 1; sh < 32; sh <<= 1) { s += __shfl_xor(s, sh, 32); s2 += __shfl_xor(s2, sh, 32); }
  float mean = s * (1.f / 128.f);
  float var = s2 * (1.f / 128.f) - mean * mean;
  float inv = rsqrtf(var + 1e-5f);
#pragma unroll
  for (int i = 0; i < 4; ++i) {
    int c = lane + i * 32;
    of[(b * NTOK + n) * CCH + c] = (bf16_t)((vals[i] - mean) * inv * g[c] + bta[c]);
  }
}

// ---- K2b: LayerNorm of short^T -> sn (bf16 [B,N,C]) -----------------------
__global__ __launch_bounds__(256) void k_ln_short(const float* __restrict__ shortbuf,
    const float* __restrict__ g, const float* __restrict__ bta, bf16_t* __restrict__ sn) {
  int idx = blockIdx.x * 8 + (threadIdx.x >> 5);
  int lane = threadIdx.x & 31;
  int b = idx >> 10, n = idx & 1023;
  float vals[4], s = 0.f, s2 = 0.f;
#pragma unroll
  for (int i = 0; i < 4; ++i) {
    int c = lane + i * 32;
    float v = shortbuf[(b * CCH + c) * NTOK + n];
    vals[i] = v; s += v; s2 += v * v;
  }
#pragma unroll
  for (int sh = 1; sh < 32; sh <<= 1) { s += __shfl_xor(s, sh, 32); s2 += __shfl_xor(s2, sh, 32); }
  float mean = s * (1.f / 128.f);
  float var = s2 * (1.f / 128.f) - mean * mean;
  float inv = rsqrtf(var + 1e-5f);
#pragma unroll
  for (int i = 0; i < 4; ++i) {
    int c = lane + i * 32;
    sn[(b * NTOK + n) * CCH + c] = (bf16_t)((vals[i] - mean) * inv * g[c] + bta[c]);
  }
}

// ---- K3: qkv GEMM [16384,128] x [384,128]^T with routed outputs -----------
__global__ __launch_bounds__(256) void k_qkv(const bf16_t* __restrict__ of,
    const bf16_t* __restrict__ wq, const float* __restrict__ qkv_b,
    bf16_t* __restrict__ qb, bf16_t* __restrict__ kb, float* __restrict__ vb) {
  int lane = threadIdx.x & 31, wid = threadIdx.x >> 5;
  int tile = blockIdx.x * 8 + wid;          // 24576 tiles
  int mt = tile % 1024, jt = tile / 1024;
  int row0 = mt * 16, j0 = jt * 16;
  __builtin_prefetch(&of[row0 * CCH], 0, 3);
  v8f acc = gemm_tile_128(&of[row0 * CCH], &wq[j0 * CCH], lane);
  int nlo = lane & 15, hi = (lane & 16) >> 1;
  int j = j0 + nlo;
  int s = j >> 7, h = (j >> 5) & 3, d = j & 31;
  float bias = qkv_b[j];
#pragma unroll
  for (int r = 0; r < 8; ++r) {
    int row = row0 + r + hi;
    int b = row >> 10, nn = row & 1023;
    float v = acc[r] + bias;
    if (s == 0)      qb[((b * NHEAD + h) * NTOK + nn) * HDIM + d] = (bf16_t)v;
    else if (s == 1) kb[((b * NHEAD + h) * NTOK + nn) * HDIM + d] = (bf16_t)v;
    else             vb[(b * NTOK + nn) * CCH + h * HDIM + d] = v;
  }
}

// ---- K4: yv GEMM; output TRANSPOSED: yvT[bh][d][pos] ----------------------
__global__ __launch_bounds__(256) void k_yv(const bf16_t* __restrict__ sn,
    const bf16_t* __restrict__ wyv, const float* __restrict__ yv_b, bf16_t* __restrict__ yvT) {
  int lane = threadIdx.x & 31, wid = threadIdx.x >> 5;
  int tile = blockIdx.x * 8 + wid;          // 8192 tiles
  int jt = tile & 7, mt = tile >> 3;
  int row0 = mt * 16, j0 = jt * 16;
  v8f acc = gemm_tile_128(&sn[row0 * CCH], &wyv[j0 * CCH], lane);
  int nlo = lane & 15, hi = (lane & 16) >> 1;
  int j = j0 + nlo;
  float bias = yv_b[j];
#pragma unroll
  for (int r = 0; r < 8; ++r) {
    int row = row0 + r + hi;
    int b = row >> 10, nn = row & 1023;
    int bh = b * NHEAD + (j >> 5);
    yvT[(bh * HDIM + (j & 31)) * NTOK + nn] = (bf16_t)(acc[r] + bias);
  }
}

// ---- K5: flash attention: softmax(qk^T/sqrt(32)) @ yv + v -> xv0 bf16 -----
__global__ __launch_bounds__(256) void k_attn(const bf16_t* __restrict__ qb,
    const bf16_t* __restrict__ kb, const bf16_t* __restrict__ yvT,
    const float* __restrict__ vb, bf16_t* __restrict__ xv0) {
  __shared__ __align__(16) bf16_t Pl[8 * 512];  // per-wave 16x32 P staging (A layout)
  int lane = threadIdx.x & 31, wid = threadIdx.x >> 5;
  int gw = blockIdx.x * 8 + wid;            // 4096 row-tiles
  int bh = gw >> 6, rt = gw & 63;
  int b = bh >> 2, h = bh & 3;
  v16bf aq = load_A(&qb[(bh * NTOK + rt * 16) * HDIM], HDIM, lane);
  v8f zero = {};
  v8f o0 = zero, o1 = zero;
  float mr[8], lr[8];
#pragma unroll
  for (int r = 0; r < 8; ++r) { mr[r] = -1e30f; lr[r] = 0.f; }
  bf16_t* P = &Pl[wid * 512];
  int nlo = lane & 15;
  const float sc = 0.17677669529663687f;    // 32^-0.5
  for (int mc = 0; mc < 32; ++mc) {         // 32 key chunks of 32
    const bf16_t* kbase = &kb[(bh * NTOK + mc * 32) * HDIM];
    v8f s0 = wmma_bf16(aq, load_BT(kbase, HDIM, lane), zero);
    v8f s1 = wmma_bf16(aq, load_BT(kbase + 16 * HDIM, HDIM, lane), zero);
#pragma unroll
    for (int r = 0; r < 8; ++r) {
      float x0 = s0[r] * sc, x1 = s1[r] * sc;
      float rm = fmaxf(x0, x1);
#pragma unroll
      for (int sh = 1; sh < 16; sh <<= 1) rm = fmaxf(rm, __shfl_xor(rm, sh, 32));
      float mnew = fmaxf(mr[r], rm);
      float fac = __expf(mr[r] - mnew);
      float p0 = __expf(x0 - mnew), p1 = __expf(x1 - mnew);
      float rs = p0 + p1;
#pragma unroll
      for (int sh = 1; sh < 16; sh <<= 1) rs += __shfl_xor(rs, sh, 32);
      lr[r] = lr[r] * fac + rs;
      mr[r] = mnew;
      o0[r] *= fac; o1[r] *= fac;
      int rowr = r + ((lane & 16) >> 1);
      P[rowr * 32 + nlo] = (bf16_t)p0;
      P[rowr * 32 + nlo + 16] = (bf16_t)p1;
    }
    v16bf ap = load_A(P, 32, lane);         // same-wave LDS transpose
    const bf16_t* yvbase = &yvT[(bh * HDIM) * NTOK + mc * 32];
    o0 = wmma_bf16(ap, load_BT(yvbase, NTOK, lane), o0);
    o1 = wmma_bf16(ap, load_BT(yvbase + 16 * NTOK, NTOK, lane), o1);
  }
  int hi = (lane & 16) >> 1;
#pragma unroll
  for (int r = 0; r < 8; ++r) {
    int n = rt * 16 + r + hi;
    float inv = 1.f / lr[r];
    int base = (b * NTOK + n) * CCH + h * HDIM;
    xv0[base + nlo]      = (bf16_t)(vb[base + nlo]      + o0[r] * inv);
    xv0[base + nlo + 16] = (bf16_t)(vb[base + nlo + 16] + o1[r] * inv);
  }
}

// ---- K6: proj GEMM + residual(ori) -> xv2f [B,C,N], xv2b [B,N,C] ----------
__global__ __launch_bounds__(256) void k_proj(const bf16_t* __restrict__ xv0,
    const bf16_t* __restrict__ wp, const float* __restrict__ pb,
    const float* __restrict__ shortbuf, float* __restrict__ xv2f, bf16_t* __restrict__ xv2b) {
  int lane = threadIdx.x & 31, wid = threadIdx.x >> 5;
  int tile = blockIdx.x * 8 + wid;          // 8192 tiles
  int jt = tile & 7, mt = tile >> 3;
  int row0 = mt * 16, j0 = jt * 16;
  __builtin_prefetch(&xv0[row0 * CCH], 0, 3);
  v8f acc = gemm_tile_128(&xv0[row0 * CCH], &wp[j0 * CCH], lane);
  int nlo = lane & 15, hi = (lane & 16) >> 1;
  int j = j0 + nlo;
  float bias = pb[j];
#pragma unroll
  for (int r = 0; r < 8; ++r) {
    int row = row0 + r + hi;
    int b = row >> 10, nn = row & 1023;
    int oi = (b * CCH + j) * NTOK + nn;
    float v = acc[r] + bias + shortbuf[oi];
    xv2f[oi] = v;
    xv2b[(b * NTOK + nn) * CCH + j] = (bf16_t)v;   // [B,N,C] for conv staging
  }
}

// ---- K7: c1d conv (5-tap shifted GEMM, M=128 K=640) + residual -> xv3 -----
__global__ __launch_bounds__(256) void k_c1d(const bf16_t* __restrict__ xv2b,
    const float* __restrict__ xv2f, const bf16_t* __restrict__ wc, float* __restrict__ xv3) {
  __shared__ __align__(16) bf16_t Xc[132 * 136];   // B^T: [col 0..131][cin], stride 136
  int tid = threadIdx.x, lane = tid & 31, wid = tid >> 5;
  int b = blockIdx.x >> 3, ch = blockIdx.x & 7;
  int n0 = ch * 128;
  __builtin_prefetch(&wc[(wid * 16) * 640], 0, 3);
  // stage transposed slab with 128-bit copies from [B,N,C] layout
  for (int t = tid; t < 132 * 16; t += 256) {
    int col = t >> 4, ci8 = (t & 15) * 8;
    int n = n0 - 2 + col;
    v8bf v = {};
    if (n >= 0 && n < NTOK) v = *(const v8bf*)&xv2b[(b * NTOK + n) * CCH + ci8];
    *(v8bf*)&Xc[col * 136 + ci8] = v;
  }
  __syncthreads();
  int m0 = wid * 16;
  v8f zero = {};
  v8f acc[8];
#pragma unroll
  for (int i = 0; i < 8; ++i) acc[i] = zero;
  for (int s = 0; s < 20; ++s) {            // K = 5 taps * 128 cin
    int tap = s >> 2, cin0 = (s & 3) * 32;
    v16bf a = load_A(&wc[m0 * 640 + tap * 128 + cin0], 640, lane);
#pragma unroll
    for (int nt = 0; nt < 8; ++nt) {
      v16bf bb = load_BT(&Xc[(nt * 16 + tap) * 136 + cin0], 136, lane);
      acc[nt] = wmma_bf16(a, bb, acc[nt]);
    }
  }
  int nlo = lane & 15, hi = (lane & 16) >> 1;
#pragma unroll
  for (int nt = 0; nt < 8; ++nt)
#pragma unroll
    for (int r = 0; r < 8; ++r) {
      int c = m0 + r + hi;
      int n = n0 + nt * 16 + nlo;
      int oi = (b * CCH + c) * NTOK + n;
      xv3[oi] = acc[nt][r] + xv2f[oi];
    }
}

// ---- K8: bilinear 32->96 align_corners -> out -----------------------------
__global__ __launch_bounds__(256) void k_bilinear(const float* __restrict__ xv3,
                                                  float* __restrict__ out) {
  int t = blockIdx.x * 256 + threadIdx.x;
  int x = t % 96, y = (t / 96) % 96, bc = t / 9216;
  float fy = y * (31.f / 95.f);
  int iy0 = (int)fy; int iy1 = iy0 + 1; if (iy1 > 31) iy1 = 31;
  float wy = fy - (float)iy0;
  float fx = x * (31.f / 95.f);
  int ix0 = (int)fx; int ix1 = ix0 + 1; if (ix1 > 31) ix1 = 31;
  float wx = fx - (float)ix0;
  const float* p = &xv3[bc * 1024];
  float v00 = p[iy0 * 32 + ix0], v01 = p[iy0 * 32 + ix1];
  float v10 = p[iy1 * 32 + ix0], v11 = p[iy1 * 32 + ix1];
  out[t] = (v00 * (1.f - wx) + v01 * wx) * (1.f - wy) +
           (v10 * (1.f - wx) + v11 * wx) * wy;
}

// ---- host launch ----------------------------------------------------------
extern "C" void kernel_launch(void* const* d_in, const int* in_sizes, int n_in,
                              void* d_out, int out_size, void* d_ws, size_t ws_size,
                              hipStream_t stream) {
  (void)in_sizes; (void)n_in; (void)out_size; (void)ws_size;
  const float* s1     = (const float*)d_in[0];
  const float* o      = (const float*)d_in[1];
  const int*   index  = (const int*)d_in[2];
  const float* pe_w   = (const float*)d_in[3];
  const float* bn_g   = (const float*)d_in[4];
  const float* bn_b   = (const float*)d_in[5];
  const float* bn_m   = (const float*)d_in[6];
  const float* bn_v   = (const float*)d_in[7];
  const float* lnx_g  = (const float*)d_in[8];
  const float* lnx_b  = (const float*)d_in[9];
  const float* lny_g  = (const float*)d_in[10];
  const float* lny_b  = (const float*)d_in[11];
  const float* qkv_w  = (const float*)d_in[12];
  const float* qkv_b  = (const float*)d_in[13];
  const float* yv_w   = (const float*)d_in[14];
  const float* yv_b   = (const float*)d_in[15];
  const float* proj_w = (const float*)d_in[16];
  const float* proj_b = (const float*)d_in[17];
  const float* c1d_w  = (const float*)d_in[18];
  float* out = (float*)d_out;

  char* ws = (char*)d_ws;
  size_t off = 0;
  auto alloc = [&](size_t bytes) -> void* {
    off = (off + 255) & ~(size_t)255;
    void* p = (void*)(ws + off);
    off += bytes;
    return p;
  };
  const size_t F32_BCN = (size_t)BATCH * CCH * NTOK * 4;   // 8 MB
  const size_t BF_BNC  = (size_t)BATCH * NTOK * CCH * 2;   // 4 MB
  float*  shortbuf = (float*) alloc(F32_BCN);
  bf16_t* ofb      = (bf16_t*)alloc(BF_BNC);
  bf16_t* snb      = (bf16_t*)alloc(BF_BNC);
  bf16_t* qb       = (bf16_t*)alloc(BF_BNC);
  bf16_t* kbuf     = (bf16_t*)alloc(BF_BNC);
  bf16_t* yvTb     = (bf16_t*)alloc(BF_BNC);
  float*  vb       = (float*) alloc(F32_BCN);
  bf16_t* xv0b     = (bf16_t*)alloc(BF_BNC);
  float*  xv2f     = (float*) alloc(F32_BCN);
  bf16_t* xv2b     = (bf16_t*)alloc(BF_BNC);
  float*  xv3      = (float*) alloc(F32_BCN);
  bf16_t* wqb      = (bf16_t*)alloc(384 * 128 * 2);
  bf16_t* wyvb     = (bf16_t*)alloc(128 * 128 * 2);
  bf16_t* wprojb   = (bf16_t*)alloc(128 * 128 * 2);
  bf16_t* wcb      = (bf16_t*)alloc(128 * 640 * 2);

  k_convert  <<<320,   256, 0, stream>>>(qkv_w, yv_w, proj_w, c1d_w, wqb, wyvb, wprojb, wcb);
  k_patch    <<<256,   256, 0, stream>>>(s1, pe_w, bn_g, bn_b, bn_m, bn_v, index, shortbuf);
  k_pool_ln_o<<<2048,  256, 0, stream>>>(o, lnx_g, lnx_b, ofb);
  k_ln_short <<<2048,  256, 0, stream>>>(shortbuf, lny_g, lny_b, snb);
  k_qkv      <<<3072,  256, 0, stream>>>(ofb, wqb, qkv_b, qb, kbuf, vb);
  k_yv       <<<1024,  256, 0, stream>>>(snb, wyvb, yv_b, yvTb);
  k_attn     <<<512,   256, 0, stream>>>(qb, kbuf, yvTb, vb, xv0b);
  k_proj     <<<1024,  256, 0, stream>>>(xv0b, wprojb, proj_b, shortbuf, xv2f, xv2b);
  k_c1d      <<<128,   256, 0, stream>>>(xv2b, xv2f, wcb, xv3);
  k_bilinear <<<73728, 256, 0, stream>>>(xv3, out);
}
